// TaylorSeriesChannelMixer_26027501814564
// MI455X (gfx1250) — matrix operand order repaired and checked by reading, not compiled, not run on hardware
//
#include <hip/hip_runtime.h>

// ---------------------------------------------------------------------------
// TaylorSeriesChannelMixer for MI455X (gfx1250, wave32, WMMA f16->f32)
// ---------------------------------------------------------------------------
// Shapes: x[B=4, C=16, H=256, W=256], HID=64, O=16 output rolls (shift 1).
// One wave handles a tile of 16 consecutive pixels (along W) for all 16 rolls.
// GEMM1: t_rolled[16x16] x W1^T[16x64]   -> 4  v_wmma_f32_16x16x32_f16 / roll
// GEMM2: relu(h1)[16x64] x W2^T[64x64]   -> 8  v_wmma_f32_16x16x32_f16 / roll
// Epilogue: h3 = W3.relu(h2+b2) via shuffles; cv branch precomputed; minmax
// via ordered-uint atomics; in-place normalize in a second kernel.
// ---------------------------------------------------------------------------

typedef __attribute__((ext_vector_type(16))) _Float16 v16h;
typedef __attribute__((ext_vector_type(8)))  _Float16 v8h;
typedef __attribute__((ext_vector_type(4)))  _Float16 v4h;
typedef __attribute__((ext_vector_type(8)))  float    v8f;

#define B_   4
#define C_   16
#define H_   256
#define W_   256
#define HW_  65536
#define HID_ 64
#define O_   16

__device__ __forceinline__ void lds_fence() {
  // intra-wave LDS write->read ordering (cross-lane through LDS)
  asm volatile("s_wait_dscnt 0" ::: "memory");
}

__device__ __forceinline__ unsigned enc_ordered(float f) {
  unsigned u = __float_as_uint(f);
  return (f >= 0.f) ? (u | 0x80000000u) : ~u;
}
__device__ __forceinline__ float dec_ordered(unsigned u) {
  return (u & 0x80000000u) ? __uint_as_float(u ^ 0x80000000u)
                           : __uint_as_float(~u);
}

__global__ void init_minmax_kernel(unsigned* gmin, unsigned* gmax) {
  int i = threadIdx.x;
  if (i < B_ * O_) { gmin[i] = 0xFFFFFFFFu; gmax[i] = 0u; }
}

__global__ __launch_bounds__(256)
void taylor_mixer_kernel(const float* __restrict__ x,
                         const float* __restrict__ coeff,
                         const float* __restrict__ wc,
                         const float* __restrict__ bc,
                         const float* __restrict__ W1,
                         const float* __restrict__ b1,
                         const float* __restrict__ W2,
                         const float* __restrict__ b2,
                         const float* __restrict__ W3,
                         const float* __restrict__ b3,
                         float* __restrict__ out,
                         unsigned* __restrict__ gmin,
                         unsigned* __restrict__ gmax) {
  __shared__ _Float16 t2[8][16][32];   // poly features, duplicated for rolls
  __shared__ _Float16 h1p[8][16][64];  // K-permuted h1 (A-operand for GEMM2)
  __shared__ float    cvs[8][16][16];  // conv1x1 branch per (pix, roll)
  __shared__ float    h3s[8][16];      // per-pixel MLP output
  __shared__ unsigned bmin[16], bmax[16];

  const int lane = threadIdx.x & 31;
  const int wave = threadIdx.x >> 5;
  const int pb   = blockIdx.x * 128 + wave * 16;  // tile base pixel
  const int b    = pb >> 16;                      // batch (block-uniform)
  const int hw   = pb & (HW_ - 1);                // offset within image

  if (threadIdx.x < 16) { bmin[threadIdx.x] = 0xFFFFFFFFu; bmax[threadIdx.x] = 0u; }
  __syncthreads();

  // ---- polynomial features ------------------------------------------------
  const float k0 = coeff[0], k1 = coeff[1];
  const float k2 = coeff[2] * 0.5f;
  const float k3 = coeff[3] * (1.f / 6.f);
  const float k4 = coeff[4] * (1.f / 24.f);

  const int pix = lane & 15;
  const int cg  = (lane >> 4) * 8;
#pragma unroll
  for (int i = 0; i < 8; ++i) {
    int c = cg + i;
    float xv = x[(b * C_ + c) * HW_ + hw + pix];
    float t  = k0 + xv * (k1 + xv * (k2 + xv * (k3 + xv * k4)));
    _Float16 th = (_Float16)t;
    t2[wave][pix][c]      = th;
    t2[wave][pix][c + 16] = th;   // duplicate: roll o = contiguous window
  }
  lds_fence();

  // ---- cv branch for all rolls (once per tile) ---------------------------
  {
    int og = (lane >> 4) * 8;
#pragma unroll
    for (int oo = 0; oo < 8; ++oo) {
      int o = og + oo;
      float s = 0.f;
#pragma unroll
      for (int c = 0; c < 16; ++c) s += wc[c] * (float)t2[wave][pix][o + c];
      cvs[wave][pix][o] = s;
    }
  }
  lds_fence();

  // ---- load weight operand registers (once per wave) ----------------------
  // B layout (f16 16x16x32): lane n%16 = column N;
  // half j -> K = 8*(lane/16) + j (j<8), 16 + 8*(lane/16) + (j-8) (j>=8).
  const int nsel = lane & 15;
  const int kgrp = (lane >> 4) * 16;

  v16h b1reg[4];  // W1^T, K padded 16->32 with zeros
#pragma unroll
  for (int nt = 0; nt < 4; ++nt)
#pragma unroll
    for (int j = 0; j < 16; ++j) {
      int k = (kgrp >> 1) + j + ((j >> 3) << 3);
      b1reg[nt][j] = (k < 16) ? (_Float16)W1[(nt * 16 + nsel) * 16 + k]
                              : (_Float16)0.f;
    }

  // W2^T with K-permutation k' = q*4 + nt_in  <->  hid = nt_in*16 + q,
  // matching the packed h1 store below (so GEMM2 is exact).
  v16h b2reg[4][2];
#pragma unroll
  for (int nt = 0; nt < 4; ++nt)
#pragma unroll
    for (int ch = 0; ch < 2; ++ch)
#pragma unroll
      for (int j = 0; j < 16; ++j) {
        int kp  = ch * 32 + (kgrp >> 1) + j + ((j >> 3) << 3);  // permuted K 0..63
        int hid = (kp & 3) * 16 + (kp >> 2);
        b2reg[nt][ch][j] = (_Float16)W2[(nt * 16 + nsel) * 64 + hid];
      }

  float b1v[4], b2v[4], w3v[4];
#pragma unroll
  for (int nt = 0; nt < 4; ++nt) {
    b1v[nt] = b1[nt * 16 + nsel];
    b2v[nt] = b2[nt * 16 + nsel];
    w3v[nt] = W3[nt * 16 + nsel];
  }
  const float cvbase = bc[0] + b3[0];

  const int g     = (lane >> 4) * 8;   // A-operand K sub-group offset
  const int M     = pix;               // A-operand row
  const int rbase = (lane >> 4) * 8;   // D-operand row base
  const int q4    = nsel * 4;          // permuted store column base

  // ---- main loop over the 16 channel rolls --------------------------------
#pragma unroll 1
  for (int o = 0; o < 16; ++o) {
    // GEMM1 A: rolled features (K=16, zero-padded to 32)
    v16h a1;
#pragma unroll
    for (int j = 0; j < 8; ++j) a1[j] = t2[wave][M][o + g + j];
#pragma unroll
    for (int j = 8; j < 16; ++j) a1[j] = (_Float16)0.f;

    v8f d1[4];
#pragma unroll
    for (int nt = 0; nt < 4; ++nt) {
      v8f z = {0.f, 0.f, 0.f, 0.f, 0.f, 0.f, 0.f, 0.f};
      d1[nt] = __builtin_amdgcn_wmma_f32_16x16x32_f16(
          false, a1, false, b1reg[nt], (short)0, z, false, false);
    }

    // bias + relu + K-permuted f16 pack: 4 tile values -> one b64 store
#pragma unroll
    for (int r = 0; r < 8; ++r) {
      v4h pk;
#pragma unroll
      for (int nt = 0; nt < 4; ++nt) {
        float v = d1[nt][r] + b1v[nt];
        pk[nt] = (_Float16)(v > 0.f ? v : 0.f);
      }
      *(v4h*)&h1p[wave][rbase + r][q4] = pk;
    }
    lds_fence();

    // GEMM2 A: two K=32 chunks, each two aligned b128 LDS reads per lane
    v16h a2[2];
#pragma unroll
    for (int ch = 0; ch < 2; ++ch) {
      v8h lo = *(const v8h*)&h1p[wave][M][ch * 32 + g];
      v8h hi = *(const v8h*)&h1p[wave][M][ch * 32 + 16 + g];
#pragma unroll
      for (int j = 0; j < 8; ++j) { a2[ch][j] = lo[j]; a2[ch][j + 8] = hi[j]; }
    }

    v8f c2[4];
#pragma unroll
    for (int nt = 0; nt < 4; ++nt) {
      v8f z = {0.f, 0.f, 0.f, 0.f, 0.f, 0.f, 0.f, 0.f};
      z = __builtin_amdgcn_wmma_f32_16x16x32_f16(
          false, a2[0], false, b2reg[nt][0], (short)0, z, false, false);
      z = __builtin_amdgcn_wmma_f32_16x16x32_f16(
          false, a2[1], false, b2reg[nt][1], (short)0, z, false, false);
      c2[nt] = z;
    }

    // h3 = W3 . relu(h2 + b2): per-lane partials + 16-lane butterfly
    float s[8];
#pragma unroll
    for (int r = 0; r < 8; ++r) {
      float acc = 0.f;
#pragma unroll
      for (int nt = 0; nt < 4; ++nt) {
        float v = c2[nt][r] + b2v[nt];
        acc += w3v[nt] * (v > 0.f ? v : 0.f);
      }
      s[r] = acc;
    }
#pragma unroll
    for (int m = 1; m < 16; m <<= 1)
#pragma unroll
      for (int r = 0; r < 8; ++r) s[r] += __shfl_xor(s[r], m, 16);

    if (lane == 0) {
#pragma unroll
      for (int r = 0; r < 8; ++r) h3s[wave][r] = s[r];
    }
    if (lane == 16) {
#pragma unroll
      for (int r = 0; r < 8; ++r) h3s[wave][8 + r] = s[r];
    }
    lds_fence();

    // epilogue: leaky-relu, store y, track ordered min/max
    unsigned u_mn = 0xFFFFFFFFu, u_mx = 0u;
    if (lane < 16) {
      float y = cvs[wave][lane][o] + h3s[wave][lane] + cvbase;
      y = (y >= 0.f) ? y : 0.1f * y;
      out[(b * O_ + o) * HW_ + hw + lane] = y;
      unsigned u = enc_ordered(y);
      u_mn = u; u_mx = u;
    }
#pragma unroll
    for (int m = 1; m < 32; m <<= 1) {
      unsigned a = __shfl_xor(u_mn, m, 32); u_mn = (a < u_mn) ? a : u_mn;
      unsigned c = __shfl_xor(u_mx, m, 32); u_mx = (c > u_mx) ? c : u_mx;
    }
    if (lane == 0) { atomicMin(&bmin[o], u_mn); atomicMax(&bmax[o], u_mx); }
  }

  __syncthreads();
  if (threadIdx.x < 16) {
    int bblk = (blockIdx.x * 128) >> 16;   // batch index (block-uniform)
    atomicMin(&gmin[bblk * O_ + threadIdx.x], bmin[threadIdx.x]);
    atomicMax(&gmax[bblk * O_ + threadIdx.x], bmax[threadIdx.x]);
  }
}

__global__ void normalize_kernel(float* __restrict__ out,
                                 const unsigned* __restrict__ gmin,
                                 const unsigned* __restrict__ gmax,
                                 int n) {
  int idx = blockIdx.x * blockDim.x + threadIdx.x;
  if (idx >= n) return;
  int bo = idx >> 16;                 // (b,o) pair: HW_ = 65536 elems each
  float mn = dec_ordered(gmin[bo]);
  float mx = dec_ordered(gmax[bo]);
  out[idx] = (out[idx] - mn) / (mx - mn + 1e-8f);
}

extern "C" void kernel_launch(void* const* d_in, const int* in_sizes, int n_in,
                              void* d_out, int out_size, void* d_ws, size_t ws_size,
                              hipStream_t stream) {
  (void)in_sizes; (void)n_in; (void)ws_size;
  const float* x     = (const float*)d_in[0];
  const float* coeff = (const float*)d_in[1];
  const float* wc    = (const float*)d_in[2];
  const float* bc    = (const float*)d_in[3];
  const float* W1    = (const float*)d_in[4];
  const float* b1    = (const float*)d_in[5];
  const float* W2    = (const float*)d_in[6];
  const float* b2    = (const float*)d_in[7];
  const float* W3    = (const float*)d_in[8];
  const float* b3    = (const float*)d_in[9];
  float* out = (float*)d_out;

  unsigned* gmin = (unsigned*)d_ws;        // 64 words
  unsigned* gmax = gmin + B_ * O_;         // 64 words

  init_minmax_kernel<<<1, 64, 0, stream>>>(gmin, gmax);

  int blocks = (B_ * HW_) / 128;           // 2048 blocks, 8 waves x 16 pixels
  taylor_mixer_kernel<<<blocks, 256, 0, stream>>>(
      x, coeff, wc, bc, W1, b1, W2, b2, W3, b3, out, gmin, gmax);

  int n = out_size;                        // 4*16*256*256
  normalize_kernel<<<(n + 255) / 256, 256, 0, stream>>>(out, gmin, gmax, n);
}
